// SequenceAttLayer_45655502356610
// MI455X (gfx1250) — matrix operand order repaired and checked by reading, not compiled, not run on hardware
//
#include <hip/hip_runtime.h>
#include <hip/hip_bf16.h>
#include <math.h>

typedef _Float16 f16;
typedef _Float16 v16h __attribute__((ext_vector_type(16)));
typedef _Float16 v8h  __attribute__((ext_vector_type(8)));
typedef _Float16 v2h  __attribute__((ext_vector_type(2)));
typedef float    v8f  __attribute__((ext_vector_type(8)));

namespace {
constexpr int Tn = 200, Hn = 64, D1 = 80, D2 = 40;
constexpr int TPAD = 208;   // 13 * 16 (M padded)
constexpr int MT   = 13;    // M tiles
constexpr int K1   = 64;    // GEMM1 K
constexpr int K2   = 96;    // GEMM2 K (80 padded to 3*32)
constexpr int N2P  = 48;    // GEMM2 N (40 padded to 3*16)

// dynamic-LDS byte offsets (all f16 regions 16B aligned for ds_load_b128)
constexpr int OFF_QS   = 0;          // 64  f32
constexpr int OFF_QWB1 = 256;        // 80  f32 (q-folded bias for layer1)
constexpr int OFF_B2S  = 576;        // 40  f32
constexpr int OFF_WDS  = 736;        // 40  f32
constexpr int OFF_RED  = 896;        // 8   f32 (cross-wave reduce)
constexpr int OFF_ATT  = 928;        // 208 f32
constexpr int OFF_PART = 1760;       // 8*64 f32 partials
constexpr int OFF_KF   = 3808;       // 208*64  f16  keys
constexpr int OFF_WKT  = 30432;      // 80*64   f16  per-batch W (N-major)
constexpr int OFF_H1   = 40672;      // 208*96  f16
constexpr int OFF_W2T  = 80608;      // 48*96   f16  W2 (N-major)
constexpr int OFF_H2   = 89824;      // 208*40  f16
constexpr int SMEM_BYTES = 106464;
}

// fast sigmoid: v_exp_f32 + v_rcp_f32 (avoid IEEE div expansion)
__device__ __forceinline__ float sigmoidf_fast(float x) {
  return __builtin_amdgcn_rcpf(1.0f + __expf(-x));
}

__global__ __launch_bounds__(256) void din_attention_kernel(
    const float* __restrict__ Q,    // [B,64]
    const float* __restrict__ Keys, // [B,200,64]
    const int*   __restrict__ KLen, // [B]
    const float* __restrict__ W1,   // [256,80]
    const float* __restrict__ B1,   // [80]
    const float* __restrict__ W2,   // [80,40]
    const float* __restrict__ B2,   // [40]
    const float* __restrict__ Wd,   // [40]
    const float* __restrict__ Bd,   // [1]
    float* __restrict__ Out)        // [B,64]
{
  extern __shared__ char smem[];
  float* qs   = (float*)(smem + OFF_QS);
  float* qWb1 = (float*)(smem + OFF_QWB1);
  float* b2s  = (float*)(smem + OFF_B2S);
  float* wds  = (float*)(smem + OFF_WDS);
  float* red  = (float*)(smem + OFF_RED);
  float* att  = (float*)(smem + OFF_ATT);
  float* part = (float*)(smem + OFF_PART);
  f16* Kf  = (f16*)(smem + OFF_KF);
  f16* WkT = (f16*)(smem + OFF_WKT);
  f16* H1  = (f16*)(smem + OFF_H1);
  f16* W2T = (f16*)(smem + OFF_W2T);
  f16* H2  = (f16*)(smem + OFF_H2);

  const int b    = blockIdx.x;
  const int tid  = threadIdx.x;
  const int lane = tid & 31;
  const int wave = tid >> 5;
  const int lo   = lane & 15;   // N / M-row selector within fragments
  const int hi   = lane >> 4;   // half-wave selector

  if (tid == 0) { __builtin_prefetch(W1, 0, 1); __builtin_prefetch(W2, 0, 1); }

  // ---------------- Phase 0a: stage q, biases, keys(f16), zero pads -------
  if (tid < Hn) qs[tid] = Q[(size_t)b * Hn + tid];
  if (tid < D2) { b2s[tid] = B2[tid]; wds[tid] = Wd[tid]; }

  const float* kb = Keys + (size_t)b * Tn * Hn;
  for (int e = tid; e < TPAD * (Hn / 4); e += 256) {
    int t = e >> 4;
    int c = (e & 15) << 2;
    f16 v0 = (f16)0, v1 = (f16)0, v2 = (f16)0, v3 = (f16)0;
    if (t < Tn) {
      const float4 kv = *(const float4*)(kb + t * Hn + c);
      v0 = (f16)kv.x; v1 = (f16)kv.y; v2 = (f16)kv.z; v3 = (f16)kv.w;
    }
    f16* dst = Kf + t * Hn + c;
    dst[0] = v0; dst[1] = v1; dst[2] = v2; dst[3] = v3;
  }
  // zero H1 K-pad columns (80..95) so GEMM2's padded K contributes 0 (not NaN)
  for (int e = tid; e < TPAD * 16; e += 256)
    H1[(e >> 4) * K2 + 80 + (e & 15)] = (f16)0;
  // W2 transposed to N-major [48][96], zero padded
  for (int e = tid; e < N2P * K2; e += 256) {
    int n = e / K2, k = e - n * K2;
    float v = (n < D2 && k < D1) ? W2[k * D2 + n] : 0.0f;
    W2T[e] = (f16)v;
  }
  __syncthreads();

  // ---------------- Phase 0b: fold q into per-batch layer-1 weights -------
  // x*W1 = q*(W1a+W1c) + k*(W1b-W1c) + (q.*k)*W1d
  for (int e = tid; e < D1 * K1; e += 256) {
    int j = e >> 6, i = e & 63;
    float v = W1[(64 + i) * D1 + j] - W1[(128 + i) * D1 + j]
            + qs[i] * W1[(192 + i) * D1 + j];
    WkT[e] = (f16)v;                    // N-major: WkT[j*64 + i]
  }
  if (tid < D1) {
    float s = B1[tid];
    for (int i = 0; i < K1; ++i)
      s += qs[i] * (W1[i * D1 + tid] + W1[(128 + i) * D1 + tid]);
    qWb1[tid] = s;
  }
  __syncthreads();

  // ---------------- GEMM1: [208x64] x [64x80] -> sigmoid -> H1 ------------
  for (int tile = wave; tile < MT * 5; tile += 8) {
    int tm = tile % MT, tn = tile / MT;
    v8f c = {};
    const f16* Arow = Kf  + (tm * 16 + lo) * K1;
    const f16* Brow = WkT + (tn * 16 + lo) * K1;
#pragma unroll
    for (int ks = 0; ks < K1; ks += 32) {
      union { v16h v; v8h h[2]; } A, Bm;
      A.h[0]  = *(const v8h*)(Arow + ks + hi * 16);       // K: kB..kB+7
      A.h[1]  = *(const v8h*)(Arow + ks + hi * 16 + 8);   // K: kB+8..kB+15
      Bm.h[0] = *(const v8h*)(Brow + ks + hi * 16);       // K: kB..kB+7
      Bm.h[1] = *(const v8h*)(Brow + ks + hi * 16 + 8);   // K: kB+8..kB+15
      c = __builtin_amdgcn_wmma_f32_16x16x32_f16(false, A.v, false, Bm.v,
                                                 (short)0, c, false, false);
    }
    int n = tn * 16 + lo;
    float add = qWb1[n];
#pragma unroll
    for (int i = 0; i < 8; ++i) {
      int m = tm * 16 + hi * 8 + i;
      H1[m * K2 + n] = (f16)sigmoidf_fast(c[i] + add);
    }
  }
  __syncthreads();

  // ---------------- GEMM2: [208x96] x [96x48] -> sigmoid -> H2 ------------
  for (int tile = wave; tile < MT * 3; tile += 8) {
    int tm = tile % MT, tn = tile / MT;
    v8f c = {};
    const f16* Arow = H1  + (tm * 16 + lo) * K2;
    const f16* Brow = W2T + (tn * 16 + lo) * K2;
#pragma unroll
    for (int ks = 0; ks < K2; ks += 32) {
      union { v16h v; v8h h[2]; } A, Bm;
      A.h[0]  = *(const v8h*)(Arow + ks + hi * 16);
      A.h[1]  = *(const v8h*)(Arow + ks + hi * 16 + 8);
      Bm.h[0] = *(const v8h*)(Brow + ks + hi * 16);
      Bm.h[1] = *(const v8h*)(Brow + ks + hi * 16 + 8);
      c = __builtin_amdgcn_wmma_f32_16x16x32_f16(false, A.v, false, Bm.v,
                                                 (short)0, c, false, false);
    }
    int n = tn * 16 + lo;
    if (n < D2) {
      float add = b2s[n];
#pragma unroll
      for (int i = 0; i < 8; ++i) {
        int m = tm * 16 + hi * 8 + i;
        H2[m * D2 + n] = (f16)sigmoidf_fast(c[i] + add);
      }
    }
  }
  __syncthreads();

  // ---------------- scores (matvec, vectorized LDS reads) + softmax -------
  const int len = KLen[b];
  float sc = -__builtin_inff();
  if (tid < Tn && tid < len) {
    float s = Bd[0];
    const v8h* hr = (const v8h*)(H2 + tid * D2);   // 80B row, 16B aligned
#pragma unroll
    for (int v = 0; v < D2 / 8; ++v) {
      v8h hv = hr[v];
#pragma unroll
      for (int j = 0; j < 8; ++j) s += (float)hv[j] * wds[v * 8 + j];
    }
    sc = s * 0.125f;            // / sqrt(64)
  }
  float mx = sc;
#pragma unroll
  for (int off = 16; off > 0; off >>= 1)
    mx = fmaxf(mx, __shfl_xor(mx, off, 32));
  if (lane == 0) red[wave] = mx;
  __syncthreads();
  mx = red[0];
#pragma unroll
  for (int w = 1; w < 8; ++w) mx = fmaxf(mx, red[w]);
  float p = (tid < Tn) ? __expf(sc - mx) : 0.0f;   // masked lanes: exp(-inf)=0
  float sm = p;
#pragma unroll
  for (int off = 16; off > 0; off >>= 1)
    sm += __shfl_xor(sm, off, 32);
  __syncthreads();
  if (lane == 0) red[wave] = sm;
  __syncthreads();
  float tot = 0.0f;
#pragma unroll
  for (int w = 0; w < 8; ++w) tot += red[w];
  float inv = __builtin_amdgcn_rcpf(tot);
  if (tid < Tn) att[tid] = p * inv;
  __syncthreads();

  // ---------------- out = att @ keys (packed f16-pair reads) --------------
  {
    int hp = (tid & 31) * 2;    // h pair: hp, hp+1
    int pp = tid >> 5;          // 0..7 partial slot
    float a0 = 0.0f, a1 = 0.0f;
    for (int t = pp; t < Tn; t += 8) {
      float w = att[t];
      v2h kk = *(const v2h*)(Kf + t * Hn + hp);
      a0 += w * (float)kk[0];
      a1 += w * (float)kk[1];
    }
    part[pp * Hn + hp]     = a0;
    part[pp * Hn + hp + 1] = a1;
  }
  __syncthreads();
  if (tid < Hn) {
    float s = 0.0f;
#pragma unroll
    for (int w = 0; w < 8; ++w) s += part[w * Hn + tid];
    Out[(size_t)b * Hn + tid] = s;
  }
}

extern "C" void kernel_launch(void* const* d_in, const int* in_sizes, int n_in,
                              void* d_out, int out_size, void* d_ws, size_t ws_size,
                              hipStream_t stream) {
  const float* Q    = (const float*)d_in[0];
  const float* Keys = (const float*)d_in[1];
  const int*   KLen = (const int*)d_in[2];
  const float* W1   = (const float*)d_in[3];
  const float* B1   = (const float*)d_in[4];
  const float* W2   = (const float*)d_in[5];
  const float* B2   = (const float*)d_in[6];
  const float* Wd   = (const float*)d_in[7];
  const float* Bd   = (const float*)d_in[8];
  float* Out = (float*)d_out;

  const int B = in_sizes[0] / Hn;   // 4096
  din_attention_kernel<<<B, 256, SMEM_BYTES, stream>>>(
      Q, Keys, KLen, W1, B1, W2, B2, Wd, Bd, Out);
}
